// PoolingPointCapsuleLayer_3126736191590
// MI455X (gfx1250) — matrix operand order repaired and checked by reading, not compiled, not run on hardware
//
#include <hip/hip_runtime.h>
#include <hip/hip_bf16.h>

// ---------------------------------------------------------------------------
// PoolingPointCapsuleLayer for MI455X (gfx1250, wave32, WMMA + TDM).
//
// Pipeline (all on `stream`):
//   k_mean_lrf     : 4x4 cov + power-iteration top eigvec per (b,p,c) -> mli
//   k_pack_w1/w2   : f32 -> f16 weights pre-swizzled into WMMA B-fragment
//                    layout (lane-major packed pairs -> contiguous b128 loads)
//   k_rotate       : qrotv(mli, points) -> f16 activations xh (2048 x 192)
//   k_gemm1        : xh @ W1 + b1 -> t1h (f16), v_wmma_f32_16x16x32_f16
//   k_gemm2_votes  : t1h @ W2 + b2, fused quaternion-normalize + sign fix +
//                    qmul(lrfs, t) + sign fix -> vote (fp32, 134 MB, L2-resident)
//   k_routing      : activation row staged to LDS via TDM tensor_load_to_lds,
//                    3 EM iterations, one wave32 per (b,p,o) group, votes in
//                    registers, shfl_xor reductions -> pose + agreement
// ---------------------------------------------------------------------------

typedef __attribute__((ext_vector_type(16))) _Float16 v16h;
typedef __attribute__((ext_vector_type(8)))  _Float16 v8h;
typedef __attribute__((ext_vector_type(8)))  float    v8f;
typedef __attribute__((ext_vector_type(4)))  unsigned v4u;
typedef __attribute__((ext_vector_type(8)))  unsigned v8u;

#define B_     8
#define P_     32
#define K_     8
#define CIN_   64
#define COUT_  64
#define NVOTE_ 512           // K_ * CIN_
#define EPS_   1e-12f

__device__ __forceinline__ float sgnf(float x) {
  return (x > 0.f) ? 1.f : ((x < 0.f) ? -1.f : 0.f);
}

__device__ __forceinline__ void qmul4(const float* q, const float* r, float* o) {
  o[0] = q[0]*r[0] - q[1]*r[1] - q[2]*r[2] - q[3]*r[3];
  o[1] = q[0]*r[1] + q[1]*r[0] + q[2]*r[3] - q[3]*r[2];
  o[2] = q[0]*r[2] - q[1]*r[3] + q[2]*r[0] + q[3]*r[1];
  o[3] = q[0]*r[3] + q[1]*r[2] - q[2]*r[1] + q[3]*r[0];
}

// Top eigenvector of a symmetric PSD 4x4 (packed upper triangle) via power
// iteration; matches _top_eigvec semantics (valid mask + sign fix).
__device__ __forceinline__ void top_eigvec4(const float m[10], bool valid, float v[4]) {
  float v0 = 0.5f, v1 = 0.5f, v2 = 0.5f, v3 = 0.5f;
  for (int i = 0; i < 32; ++i) {
    float w0 = m[0]*v0 + m[1]*v1 + m[2]*v2 + m[3]*v3;
    float w1 = m[1]*v0 + m[4]*v1 + m[5]*v2 + m[6]*v3;
    float w2 = m[2]*v0 + m[5]*v1 + m[7]*v2 + m[8]*v3;
    float w3 = m[3]*v0 + m[6]*v1 + m[8]*v2 + m[9]*v3;
    float nn  = sqrtf(w0*w0 + w1*w1 + w2*w2 + w3*w3);
    float inv = 1.0f / fmaxf(nn, 1e-30f);
    v0 = w0*inv; v1 = w1*inv; v2 = w2*inv; v3 = w3*inv;
  }
  if (!valid) { v0 = v1 = v2 = v3 = 0.f; }
  float s = sgnf(v0);
  v[0] = v0*s; v[1] = v1*s; v[2] = v2*s; v[3] = v3*s;
}

__device__ __forceinline__ float wave_sum(float x) {
  #pragma unroll
  for (int off = 16; off > 0; off >>= 1) x += __shfl_xor(x, off, 32);
  return x;
}

__device__ __forceinline__ unsigned pack2h(float lo, float hi) {
  _Float16 l = (_Float16)lo, h = (_Float16)hi;
  unsigned short lb = __builtin_bit_cast(unsigned short, l);
  unsigned short hb = __builtin_bit_cast(unsigned short, h);
  return (unsigned)lb | ((unsigned)hb << 16);
}

// -------------------------------------------------------------- mean LRF ---
__global__ void k_mean_lrf(const float* __restrict__ lrfs,
                           const float* __restrict__ act,
                           float* __restrict__ mli) {
  int idx = blockIdx.x * blockDim.x + threadIdx.x;
  if (idx >= B_ * P_ * CIN_) return;
  int c  = idx & (CIN_ - 1);
  int bp = idx >> 6;
  float cov[10];
  #pragma unroll
  for (int i = 0; i < 10; ++i) cov[i] = 0.f;
  float asum = 0.f;
  for (int k = 0; k < K_; ++k) {
    const float* q = lrfs + ((size_t)(bp * K_ + k) * CIN_ + c) * 4;
    float q0 = q[0], q1 = q[1], q2 = q[2], q3 = q[3];
    cov[0] += q0*q0; cov[1] += q0*q1; cov[2] += q0*q2; cov[3] += q0*q3;
    cov[4] += q1*q1; cov[5] += q1*q2; cov[6] += q1*q3;
    cov[7] += q2*q2; cov[8] += q2*q3; cov[9] += q3*q3;
    asum += fabsf(act[(size_t)(bp * K_ + k) * CIN_ + c]);
  }
  const float invK = 1.0f / (float)K_;
  #pragma unroll
  for (int i = 0; i < 10; ++i) cov[i] *= invK;
  cov[0] += 1e-6f; cov[4] += 1e-6f; cov[7] += 1e-6f; cov[9] += 1e-6f;
  float m[4];
  top_eigvec4(cov, asum != 0.f, m);
  float* o = mli + (size_t)idx * 4;          // store conjugate (mean_lrf_inv)
  o[0] = m[0]; o[1] = -m[1]; o[2] = -m[2]; o[3] = -m[3];
}

// ------------------------------------------------- rotate points (f16 A) ---
__global__ void k_rotate(const float* __restrict__ points,
                         const float* __restrict__ mli,
                         _Float16* __restrict__ xh) {
  int idx = blockIdx.x * blockDim.x + threadIdx.x;
  if (idx >= B_ * P_ * K_ * CIN_) return;
  int c   = idx & (CIN_ - 1);
  int bpk = idx >> 6;
  int bp  = bpk >> 3;
  const float* q = mli + ((size_t)bp * CIN_ + c) * 4;
  float qq[4] = { q[0], q[1], q[2], q[3] };
  const float* p = points + (size_t)bpk * 3;
  float v4[4] = { 0.f, p[0], p[1], p[2] };
  float t[4], qc[4] = { qq[0], -qq[1], -qq[2], -qq[3] }, r[4];
  qmul4(qq, v4, t);
  qmul4(t, qc, r);
  _Float16* o = xh + (size_t)bpk * (3 * CIN_) + c * 3;
  o[0] = (_Float16)r[1]; o[1] = (_Float16)r[2]; o[2] = (_Float16)r[3];
}

// ---------------------------------------- weight pre-swizzle (B fragment) ---
// B fragment layout (16-bit 32x16): VGPR v, lanes 0-15: K=2v+8*(v/4),+1 (N=lane);
// lanes 16-31: K=8+2v+8*(v/4),+1.  Packed dword = {hi:K+1, lo:K}.
__global__ void k_pack_w1(const float* __restrict__ W1, unsigned* __restrict__ w1p) {
  int idx = blockIdx.x * blockDim.x + threadIdx.x;
  if (idx >= 4 * 6 * 256) return;
  int vv = idx & 7, lane = (idx >> 3) & 31, rest = idx >> 8;
  int kstep = rest % 6, ct = rest / 6;
  int col = ct * 16 + (lane & 15);
  int k0  = kstep * 32 + (lane >> 4) * 8 + 2 * vv + (vv >> 2) * 8;
  w1p[idx] = pack2h(W1[(size_t)k0 * COUT_ + col], W1[(size_t)(k0 + 1) * COUT_ + col]);
}

__global__ void k_pack_w2(const float* __restrict__ W2, unsigned* __restrict__ w2p) {
  int idx = blockIdx.x * blockDim.x + threadIdx.x;
  if (idx >= 4 * CIN_ * 4 * 2 * 256) return;
  int vv = idx & 7, lane = (idx >> 3) & 31;
  int rest = idx >> 8;
  int kstep = rest & 1;  rest >>= 1;
  int ct    = rest & 3;  rest >>= 2;
  int cin   = rest & 63; int comp = rest >> 6;
  int col = comp * (CIN_ * COUT_) + cin * COUT_ + ct * 16 + (lane & 15);
  int k0  = kstep * 32 + (lane >> 4) * 8 + 2 * vv + (vv >> 2) * 8;
  const size_t ldb = 4 * CIN_ * COUT_;   // 16384
  w2p[idx] = pack2h(W2[(size_t)k0 * ldb + col], W2[(size_t)(k0 + 1) * ldb + col]);
}

// --------------------------------------------------------------- GEMM 1 ----
// x(2048x192) @ W1(192x64) + b1 -> t1h.  One wave per 16x16 tile, 6 k-steps.
__global__ void __launch_bounds__(256)
k_gemm1(const _Float16* __restrict__ xh, const unsigned* __restrict__ w1p,
        const float* __restrict__ b1, _Float16* __restrict__ t1h) {
  int wid  = blockIdx.x * 8 + (threadIdx.x >> 5);
  int lane = threadIdx.x & 31;
  int mt = wid >> 2, ct = wid & 3;
  int hf = lane >> 4, n = lane & 15;
  int row = mt * 16 + (lane & 15);        // A: lanes 0-15 / 16-31 both M=0..15
  v8f acc = {};
  #pragma unroll
  for (int kstep = 0; kstep < 6; ++kstep) {
    const _Float16* abase = xh + (size_t)row * 192 + kstep * 32 + hf * 8;
    v8h alo = *(const v8h*)abase;
    v8h ahi = *(const v8h*)(abase + 16);
    v16h a;
    #pragma unroll
    for (int i = 0; i < 8; ++i) { a[i] = alo[i]; a[i + 8] = ahi[i]; }
    v16h bfrag = *(const v16h*)(w1p + (size_t)(ct * 6 + kstep) * 256 + lane * 8);
    acc = __builtin_amdgcn_wmma_f32_16x16x32_f16(false, a, false, bfrag,
                                                 (short)0, acc, false, false);
  }
  float bias = b1[ct * 16 + n];
  #pragma unroll
  for (int r = 0; r < 8; ++r) {
    int m = r + 8 * hf;                    // C layout: VGPR r -> M=r / r+8
    t1h[(size_t)(mt * 16 + m) * COUT_ + ct * 16 + n] = (_Float16)(acc[r] + bias);
  }
}

// ----------------------------------------------- GEMM 2 + fused vote math ---
// 4 accumulators (quaternion components) share one A fragment; normalize over
// the quaternion axis in-register, sign fix, qmul with lrfs, store fp32 vote.
__global__ void __launch_bounds__(256)
k_gemm2_votes(const _Float16* __restrict__ t1h, const unsigned* __restrict__ w2p,
              const float* __restrict__ b2, const float* __restrict__ lrfs,
              float* __restrict__ vote) {
  int wid  = blockIdx.x * 8 + (threadIdx.x >> 5);
  int lane = threadIdx.x & 31;
  int ct  = wid & 3;
  int cin = (wid >> 2) & 63;
  int mt  = wid >> 8;
  int hf = lane >> 4, n = lane & 15;
  int row = mt * 16 + (lane & 15);
  v8f acc[4] = { {}, {}, {}, {} };
  #pragma unroll
  for (int kstep = 0; kstep < 2; ++kstep) {
    const _Float16* abase = t1h + (size_t)row * COUT_ + kstep * 32 + hf * 8;
    v8h alo = *(const v8h*)abase;
    v8h ahi = *(const v8h*)(abase + 16);
    v16h a;
    #pragma unroll
    for (int i = 0; i < 8; ++i) { a[i] = alo[i]; a[i + 8] = ahi[i]; }
    #pragma unroll
    for (int comp = 0; comp < 4; ++comp) {
      size_t fb = ((size_t)((comp * CIN_ + cin) * 4 + ct) * 2 + kstep) * 256 + lane * 8;
      __builtin_prefetch(w2p + fb + 512, 0, 1);   // stream next fragment (L2)
      v16h bfrag = *(const v16h*)(w2p + fb);
      acc[comp] = __builtin_amdgcn_wmma_f32_16x16x32_f16(false, a, false, bfrag,
                                                         (short)0, acc[comp],
                                                         false, false);
    }
  }
  float bias[4];
  #pragma unroll
  for (int comp = 0; comp < 4; ++comp)
    bias[comp] = b2[comp * (CIN_ * COUT_) + cin * COUT_ + ct * 16 + n];
  int cout = ct * 16 + n;
  #pragma unroll
  for (int r = 0; r < 8; ++r) {
    int m   = r + 8 * hf;
    int bpk = mt * 16 + m;
    float t[4];
    #pragma unroll
    for (int comp = 0; comp < 4; ++comp) t[comp] = acc[comp][r] + bias[comp];
    float nn  = sqrtf(t[0]*t[0] + t[1]*t[1] + t[2]*t[2] + t[3]*t[3]);
    float inv = 1.0f / fmaxf(nn, EPS_);
    float s   = sgnf(t[0] * inv);
    #pragma unroll
    for (int comp = 0; comp < 4; ++comp) t[comp] *= inv * s;
    const float* q = lrfs + ((size_t)bpk * CIN_ + cin) * 4;
    float qq[4] = { q[0], q[1], q[2], q[3] };
    float v[4];
    qmul4(qq, t, v);
    float s2 = sgnf(v[0]);
    int bp = bpk >> 3, kk = bpk & 7;
    int nidx = kk * CIN_ + cin;
    float4 outv = make_float4(v[0]*s2, v[1]*s2, v[2]*s2, v[3]*s2);
    *(float4*)(vote + ((size_t)(bp * COUT_ + cout) * NVOTE_ + nidx) * 4) = outv;
  }
}

// --------------------------------------------------------------- routing ---
// One wave32 per (b,p,o): 16 votes/lane held in registers across all 3 EM
// iterations.  The shared activation row (512 f32 = 2 KB) is staged into LDS
// by the Tensor Data Mover: wave 0 issues one tensor_load_to_lds with a D#
// built per CDNA5 ISA (count=1, type=2, data_size=4B,
// tensor_dim0 = tile_dim0 = 512, no multicast), waits on TENSORcnt, and the
// workgroup barrier publishes the tile to the other 7 waves.
__global__ void __launch_bounds__(256)
k_routing(const float* __restrict__ vote, const float* __restrict__ act,
          const float* __restrict__ alpha, const float* __restrict__ beta,
          float* __restrict__ out) {
  __shared__ float s_act[NVOTE_];
  int wid  = blockIdx.x * 8 + (threadIdx.x >> 5);
  int lane = threadIdx.x & 31;
  int bp = blockIdx.x >> 3;          // uniform: identical for all 8 waves
  int o  = wid & 63;

  if (threadIdx.x < 32) {            // one TDM DMA per block (wave 0 issues)
    unsigned lds_off = (unsigned)(unsigned long long)(void*)s_act; // addr[31:0]
    unsigned long long gaddr =
        (unsigned long long)(const void*)(act + (size_t)bp * NVOTE_);
    v4u g0;
    g0[0] = 1u;                                  // count=1, user descriptor
    g0[1] = lds_off;                             // lds_addr (bytes)
    g0[2] = (unsigned)gaddr;                     // global_addr[31:0]
    g0[3] = (unsigned)((gaddr >> 32) & 0x01FFFFFFu) | (2u << 30); // [56:32]|type=2
    v8u g1;
    g1[0] = 2u << 16;                            // data_size=4B, wg_mask=0
    g1[1] = (NVOTE_ & 0xFFFFu) << 16;            // tensor_dim0[15:0]
    g1[2] = (NVOTE_ >> 16) | (1u << 16);         // tensor_dim0[31:16], dim1=1
    g1[3] = (unsigned)NVOTE_ << 16;              // tile_dim0 = 512
    g1[4] = 1u;                                  // tile_dim1=1, tile_dim2=0
    g1[5] = NVOTE_;                              // tensor_dim0_stride[31:0]
    g1[6] = 0u;                                  // stride hi / dim1_stride lo
    g1[7] = 0u;                                  // tensor_dim1_stride hi
    asm volatile("tensor_load_to_lds %0, %1" :: "s"(g0), "s"(g1) : "memory");
    __builtin_amdgcn_s_wait_tensorcnt(0);
  }
  __syncthreads();

  float vv[16][4], ae[16], bij[16];
  float totsum = 0.f;
  const float* vbase = vote + (size_t)(bp * COUT_ + o) * NVOTE_ * 4;
  #pragma unroll
  for (int j = 0; j < 16; ++j) {
    int nn = lane + j * 32;
    float4 q = *(const float4*)(vbase + (size_t)nn * 4);
    vv[j][0] = q.x; vv[j][1] = q.y; vv[j][2] = q.z; vv[j][3] = q.w;
    totsum += q.x + q.y + q.z + q.w;
    ae[j]  = s_act[nn];
    bij[j] = ae[j];
  }
  totsum = wave_sum(totsum);
  bool valid = (totsum != 0.f);
  float pose[4] = { 0.f, 0.f, 0.f, 0.f };

  for (int it = 0; it < 3; ++it) {
    float w[16], wsum = 0.f;
    #pragma unroll
    for (int j = 0; j < 16; ++j) {
      w[j] = bij[j] * ((ae[j] != 0.f) ? 1.f : 0.f);
      wsum += fabsf(w[j]);
    }
    wsum = wave_sum(wsum);
    float winv = 1.0f / fmaxf(wsum, EPS_);
    float cov[10];
    #pragma unroll
    for (int i = 0; i < 10; ++i) cov[i] = 0.f;
    #pragma unroll
    for (int j = 0; j < 16; ++j) {
      float ww = w[j] * winv;
      float a0 = vv[j][0], a1 = vv[j][1], a2 = vv[j][2], a3 = vv[j][3];
      cov[0] += ww*a0*a0; cov[1] += ww*a0*a1; cov[2] += ww*a0*a2; cov[3] += ww*a0*a3;
      cov[4] += ww*a1*a1; cov[5] += ww*a1*a2; cov[6] += ww*a1*a3;
      cov[7] += ww*a2*a2; cov[8] += ww*a2*a3; cov[9] += ww*a3*a3;
    }
    #pragma unroll
    for (int i = 0; i < 10; ++i) cov[i] = wave_sum(cov[i]);
    cov[0] += 1e-6f; cov[4] += 1e-6f; cov[7] += 1e-6f; cov[9] += 1e-6f;
    top_eigvec4(cov, valid, pose);
    if (it < 2) {
      #pragma unroll
      for (int j = 0; j < 16; ++j) {
        float d = fabsf(pose[0]*vv[j][0] + pose[1]*vv[j][1] +
                        pose[2]*vv[j][2] + pose[3]*vv[j][3]);
        float dist = 2.0f * acosf(fminf(d, 0.9999f)) * 0.31830988618f;
        bij[j] = bij[j] * ae[j] * (1.0f - dist);
      }
    }
  }

  float negsum = 0.f, wcnt = 0.f;
  #pragma unroll
  for (int j = 0; j < 16; ++j) {
    float d = fabsf(pose[0]*vv[j][0] + pose[1]*vv[j][1] +
                    pose[2]*vv[j][2] + pose[3]*vv[j][3]);
    float dist = 2.0f * acosf(fminf(d, 0.9999f)) * 0.31830988618f;
    negsum += bij[j] * (1.0f - dist);
    wcnt   += (bij[j] != 0.f) ? 1.f : 0.f;
  }
  negsum = wave_sum(negsum);
  wcnt   = wave_sum(wcnt);
  float mask = (wcnt != 0.f) ? 1.f : 0.f;
  float neg  = negsum / (wcnt + (1.0f - mask)) * mask;
  float ag   = (1.0f / (1.0f + expf(-(alpha[0]*neg + beta[0] - 1.0f)))) * mask;
  if (lane == 0) {
    float* pd = out + (size_t)(bp * COUT_ + o) * 4;
    pd[0] = pose[0]; pd[1] = pose[1]; pd[2] = pose[2]; pd[3] = pose[3];
    out[(size_t)B_ * P_ * COUT_ * 4 + bp * COUT_ + o] = ag;
  }
}

// ------------------------------------------------------------------ launch ---
extern "C" void kernel_launch(void* const* d_in, const int* in_sizes, int n_in,
                              void* d_out, int out_size, void* d_ws, size_t ws_size,
                              hipStream_t stream) {
  (void)in_sizes; (void)n_in; (void)out_size; (void)ws_size;
  const float* points = (const float*)d_in[0];
  const float* lrfs   = (const float*)d_in[1];
  const float* act    = (const float*)d_in[2];
  const float* W1     = (const float*)d_in[3];
  const float* b1     = (const float*)d_in[4];
  const float* W2     = (const float*)d_in[5];
  const float* b2     = (const float*)d_in[6];
  const float* alpha  = (const float*)d_in[7];
  const float* beta   = (const float*)d_in[8];
  float* out = (float*)d_out;

  // Workspace carve-up (~137.5 MB total; all regions 16B-aligned):
  float*    mli = (float*)d_ws;                       //  65536 f32
  _Float16* xh  = (_Float16*)(mli + 65536);           // 2048*192 f16
  unsigned* w1p = (unsigned*)(xh + 2048 * 192);       //   6144 u32
  _Float16* t1h = (_Float16*)(w1p + 6144);            // 2048*64 f16
  unsigned* w2p = (unsigned*)(t1h + 2048 * 64);       // 524288 u32
  float*    vote = (float*)(w2p + 524288);            // 33554432 f32 (134 MB)

  k_mean_lrf   <<<  64, 256, 0, stream>>>(lrfs, act, mli);
  k_pack_w1    <<<  24, 256, 0, stream>>>(W1, w1p);
  k_pack_w2    <<<2048, 256, 0, stream>>>(W2, w2p);
  k_rotate     <<< 512, 256, 0, stream>>>(points, mli, xh);
  k_gemm1      <<<  64, 256, 0, stream>>>(xh, w1p, b1, t1h);
  k_gemm2_votes<<<4096, 256, 0, stream>>>(t1h, w2p, b2, lrfs, vote);
  k_routing    <<<2048, 256, 0, stream>>>(vote, act, alpha, beta, out);
}
